// VGG_SMALL_1W1A_29343216566657
// MI455X (gfx1250) — 8-bit matrix operand order repaired and checked by reading, compiled, not run on hardware
//
#include <hip/hip_runtime.h>
#include <stdint.h>

typedef __attribute__((ext_vector_type(8))) int v8i;

#define TM 128
#define TN 128
#define THREADS 256

#if defined(__has_builtin)
#if __has_builtin(__builtin_amdgcn_global_load_async_to_lds_b128) && \
    __has_builtin(__builtin_amdgcn_s_wait_asynccnt)
#define USE_ASYNC_LDS 1
#endif
#endif
#ifndef USE_ASYNC_LDS
#define USE_ASYNC_LDS 0
#endif

#if USE_ASYNC_LDS
typedef __attribute__((vector_size(16))) int vi4_t;
typedef vi4_t __attribute__((address_space(1))) as1_vi4;
typedef vi4_t __attribute__((address_space(3))) as3_vi4;
#endif

__global__ __launch_bounds__(THREADS) void bconv_wmma_kernel(
    const int8_t* __restrict__ in, int8_t* __restrict__ out,
    const int8_t* __restrict__ bw, const float* __restrict__ scl,
    const float* __restrict__ bia,
    int H, int W, int ICp, int OC, int OICp, int pool) {
  __shared__ int cint[TM * TN];

  const int t  = threadIdx.x;
  const int m0 = blockIdx.x * TM;
  const int n0 = blockIdx.y * TN;

  const int lr = t >> 1;
  const int lh = t & 1;
  const int HW = H * W;
  const int gm = m0 + lr;
  const int nImg = gm / HW;
  const int rm   = gm - nImg * HW;
  const int oh   = rm / W;
  const int ow   = rm - oh * W;
  const int Wp2  = W + 2;
  const size_t inRowBase = ((size_t)(nImg * (H + 2) + oh) * Wp2 + ow) * (size_t)ICp;
  const size_t bwRowBase = (size_t)(n0 + lr) * (size_t)9 * (size_t)ICp;

  const int wave = t >> 5;
  const int lane = t & 31;
  const int l16  = lane & 15;
  const int h2   = lane >> 4;
  const int wm   = (wave & 1) * 64;
  const int wn   = (wave >> 1) * 32;

  v8i acc[4][2];
  v8i zero;
#pragma unroll
  for (int r = 0; r < 8; r++) zero[r] = 0;
#pragma unroll
  for (int mi = 0; mi < 4; mi++)
#pragma unroll
    for (int ni = 0; ni < 2; ni++) acc[mi][ni] = zero;

  const int nCB = ICp >> 6;
  const int S   = 9 * nCB;

  auto stage = [&](int s, int buf) {
    const int kk = s / nCB;
    const int cb = s - kk * nCB;
    const int kh = kk / 3, kw = kk - kh * 3;
    const int8_t* gA = in + inRowBase +
                       ((size_t)kh * Wp2 + kw) * (size_t)ICp +
                       ((size_t)cb << 6) + (size_t)(lh * 32);
    const int8_t* gB = bw + bwRowBase + (size_t)kk * ICp +
                       ((size_t)cb << 6) + (size_t)(lh * 32);
    int8_t* lA = (int8_t*)cint + buf * 16384 + lr * 64 + lh * 32;
    int8_t* lB = lA + 8192;
#if USE_ASYNC_LDS
    __builtin_amdgcn_global_load_async_to_lds_b128((as1_vi4*)gA,
                                                   (as3_vi4*)lA, 0, 0);
    __builtin_amdgcn_global_load_async_to_lds_b128((as1_vi4*)(gA + 16),
                                                   (as3_vi4*)(lA + 16), 0, 0);
    __builtin_amdgcn_global_load_async_to_lds_b128((as1_vi4*)gB,
                                                   (as3_vi4*)lB, 0, 0);
    __builtin_amdgcn_global_load_async_to_lds_b128((as1_vi4*)(gB + 16),
                                                   (as3_vi4*)(lB + 16), 0, 0);
#else
    int4 a0 = ((const int4*)gA)[0], a1 = ((const int4*)gA)[1];
    int4 b0 = ((const int4*)gB)[0], b1 = ((const int4*)gB)[1];
    ((int4*)lA)[0] = a0; ((int4*)lA)[1] = a1;
    ((int4*)lB)[0] = b0; ((int4*)lB)[1] = b1;
#endif
  };

  stage(0, 0);
  for (int s = 0; s < S; s++) {
    const int buf = s & 1;
#if USE_ASYNC_LDS
    __builtin_amdgcn_s_wait_asynccnt(0);
#endif
    __syncthreads();
    if (s + 1 < S) stage(s + 1, buf ^ 1);

    const int8_t* As = (const int8_t*)cint + buf * 16384;
    const int8_t* Bs = As + 8192;

    v8i bf[2];
#pragma unroll
    for (int ni = 0; ni < 2; ni++) {
      const int8_t* bp = Bs + (wn + ni * 16 + l16) * 64;
      union { v8i v; int4 q[2]; } ub;
      ub.q[0] = *(const int4*)(bp + h2 * 16);
      ub.q[1] = *(const int4*)(bp + 32 + h2 * 16);
      bf[ni] = ub.v;
    }
#pragma unroll
    for (int mi = 0; mi < 4; mi++) {
      const int8_t* ap = As + (wm + mi * 16 + l16) * 64 + h2 * 16;
      union { v8i v; long long d[4]; } ua;
      ua.d[0] = *(const long long*)(ap);
      ua.d[1] = *(const long long*)(ap + 8);
      ua.d[2] = *(const long long*)(ap + 32);
      ua.d[3] = *(const long long*)(ap + 40);
#pragma unroll
      for (int ni = 0; ni < 2; ni++) {
        acc[mi][ni] = __builtin_amdgcn_wmma_i32_16x16x64_iu8(
            true, ua.v, true, bf[ni], acc[mi][ni], false, false);
      }
    }
  }

  __syncthreads();
#pragma unroll
  for (int mi = 0; mi < 4; mi++)
#pragma unroll
    for (int ni = 0; ni < 2; ni++)
#pragma unroll
      for (int r = 0; r < 8; r++)
        cint[(wm + mi * 16 + h2 * 8 + r) * TN + (wn + ni * 16 + l16)] =
            acc[mi][ni][r];
  __syncthreads();

  if (!pool) {
    for (int idx = t; idx < TM * TN; idx += THREADS) {
      const int Ml = idx >> 7, Nl = idx & 127;
      const int oc = n0 + Nl;
      if (oc >= OC) continue;
      const int g  = m0 + Ml;
      const int n2 = g / HW;
      const int r2 = g - n2 * HW;
      const int y  = r2 / W;
      const int x  = r2 - y * W;
      const float f = (float)cint[idx] * scl[oc] + bia[oc];
      out[(((size_t)n2 * (H + 2) + y + 1) * (W + 2) + (x + 1)) * (size_t)OICp + oc] =
          f >= 0.0f ? (int8_t)1 : (int8_t)-1;
    }
  } else {
    const int pcW = W >> 1;
    for (int idx = t; idx < (TM / 4) * TN; idx += THREADS) {
      const int pm = idx >> 7, Nl = idx & 127;
      const int oc = n0 + Nl;
      if (oc >= OC) continue;
      const int pr = pm / pcW, pc = pm - pr * pcW;
      const int mA = pr * 2 * W + pc * 2;
      int z = cint[mA * TN + Nl];
      int z1 = cint[(mA + 1) * TN + Nl];     z = z1 > z ? z1 : z;
      int z2 = cint[(mA + W) * TN + Nl];     z = z2 > z ? z2 : z;
      int z3 = cint[(mA + W + 1) * TN + Nl]; z = z3 > z ? z3 : z;
      const int g  = m0 + mA;
      const int n2 = g / HW;
      const int r2 = g - n2 * HW;
      const int y  = (r2 / W) >> 1;
      const int x  = (r2 - (r2 / W) * W) >> 1;
      const int Hp = H >> 1, Wp = W >> 1;
      const float f = (float)z * scl[oc] + bia[oc];
      out[(((size_t)n2 * (Hp + 2) + y + 1) * (Wp + 2) + (x + 1)) * (size_t)OICp + oc] =
          f >= 0.0f ? (int8_t)1 : (int8_t)-1;
    }
  }
}

__global__ void prep_input_kernel(const float* __restrict__ x,
                                  int8_t* __restrict__ out) {
  const int i = blockIdx.x * blockDim.x + threadIdx.x;
  if (i >= 256 * 32 * 32) return;
  const int n = i >> 10, r = i & 1023, h = r >> 5, w = r & 31;
  const size_t base = (((size_t)n * 34 + h + 1) * 34 + (w + 1)) * 64;
  for (int c = 0; c < 3; c++) {
    const float v = x[(((size_t)n * 3 + c) * 32 + h) * 32 + w];
    out[base + c] = v >= 0.0f ? (int8_t)1 : (int8_t)-1;
  }
}

__global__ void prep_w_kernel(const float* __restrict__ w,
                              int8_t* __restrict__ bw, int OC, int IC,
                              int ICp) {
  const int i = blockIdx.x * blockDim.x + threadIdx.x;
  if (i >= OC * IC * 9) return;
  const int oc = i / (IC * 9);
  const int r  = i - oc * IC * 9;
  const int ic = r / 9;
  const int kk = r - ic * 9;
  bw[((size_t)oc * 9 + kk) * ICp + ic] = w[i] >= 0.0f ? (int8_t)1 : (int8_t)-1;
}

__global__ void prep_bn_kernel(const float* __restrict__ p,
                               float* __restrict__ scl,
                               float* __restrict__ bia, int C) {
  const int c = blockIdx.x * blockDim.x + threadIdx.x;
  if (c >= C) return;
  const float g = p[c], b = p[C + c], m = p[2 * C + c], v = p[3 * C + c];
  const float s = g * rsqrtf(v + 1e-5f);
  scl[c] = s;
  bia[c] = b - m * s;
}

__global__ __launch_bounds__(128) void fc_kernel(
    const int8_t* __restrict__ a5, const float* __restrict__ fcw,
    const float* __restrict__ fcb, float* __restrict__ out) {
  __shared__ int red[128];
  const int b = blockIdx.x, j = blockIdx.y, t = threadIdx.x;
  int acc = 0;
  for (int k = t; k < 15568; k += 128) {
    const int c = k >> 4, r = k & 15, h = r >> 2, w = r & 3;
    const int sx = a5[(((size_t)b * 6 + h + 1) * 6 + (w + 1)) * 1024 + c];
    acc += fcw[(size_t)j * 15568 + k] >= 0.0f ? sx : -sx;
  }
  red[t] = acc;
  __syncthreads();
  for (int s = 64; s > 0; s >>= 1) {
    if (t < s) red[t] += red[t + s];
    __syncthreads();
  }
  if (t == 0) out[b * 10 + j] = (float)red[0] + fcb[j];
}

extern "C" void kernel_launch(void* const* d_in, const int* in_sizes, int n_in,
                              void* d_out, int out_size, void* d_ws,
                              size_t ws_size, hipStream_t stream) {
  (void)in_sizes; (void)n_in; (void)out_size;
  const float* x   = (const float*)d_in[0];
  const float* wts[6];
  const float* ps[6];
  for (int l = 0; l < 6; l++) { wts[l] = (const float*)d_in[1 + 2 * l];
                                ps[l]  = (const float*)d_in[2 + 2 * l]; }
  const float* fcw = (const float*)d_in[13];
  const float* fcb = (const float*)d_in[14];
  float* outp = (float*)d_out;

  static const int L_H[6]    = {32, 32, 16, 16, 8, 8};
  static const int L_W[6]    = {32, 32, 16, 16, 8, 8};
  static const int L_IC[6]   = {3, 344, 352, 686, 679, 1246};
  static const int L_ICp[6]  = {64, 384, 384, 704, 704, 1280};
  static const int L_OC[6]   = {344, 352, 686, 679, 1246, 973};
  static const int L_OICp[6] = {384, 384, 704, 704, 1280, 1024};
  static const int L_OCp[6]  = {384, 384, 768, 768, 1280, 1024};
  static const int L_POOL[6] = {0, 1, 0, 1, 0, 1};
  static const size_t L_OUTSZ[6] = {113639424ull, 31850496ull, 58392576ull,
                                    18022400ull,  32768000ull, 9437184ull};

  char* ws = (char*)d_ws;
  size_t woff[6], cur = 0;
  for (int l = 0; l < 6; l++) {
    woff[l] = cur;
    cur += (size_t)L_OCp[l] * 9 * (size_t)L_ICp[l];
  }
  const size_t wTotal = cur;
  size_t soff[6], sc = 0;
  for (int l = 0; l < 6; l++) { soff[l] = sc; sc += (size_t)L_OCp[l]; }
  float* sclBase = (float*)(ws + wTotal);
  float* biaBase = sclBase + sc;
  size_t actBase = wTotal + 2 * sc * sizeof(float);
  actBase = (actBase + 255) & ~(size_t)255;
  int8_t* bufA = (int8_t*)(ws + actBase);
  int8_t* bufB = bufA + 31850496ull;
  const size_t needed = actBase + 31850496ull + 113639424ull;
  if (ws_size < needed) return;

  (void)hipMemsetAsync(ws, 0, actBase, stream);
  for (int l = 0; l < 6; l++) {
    const int nw = L_OC[l] * L_IC[l] * 9;
    prep_w_kernel<<<(nw + 255) / 256, 256, 0, stream>>>(
        wts[l], (int8_t*)(ws + woff[l]), L_OC[l], L_IC[l], L_ICp[l]);
    prep_bn_kernel<<<(L_OC[l] + 255) / 256, 256, 0, stream>>>(
        ps[l], sclBase + soff[l], biaBase + soff[l], L_OC[l]);
  }

  (void)hipMemsetAsync(bufA, 0, 18939904ull, stream);
  prep_input_kernel<<<(256 * 1024 + 255) / 256, 256, 0, stream>>>(x, bufA);

  for (int l = 0; l < 6; l++) {
    int8_t* src = (l & 1) ? bufB : bufA;
    int8_t* dst = (l & 1) ? bufA : bufB;
    (void)hipMemsetAsync(dst, 0, L_OUTSZ[l], stream);
    const int Mtot = 256 * L_H[l] * L_W[l];
    dim3 grid(Mtot / TM, L_OCp[l] / TN);
    bconv_wmma_kernel<<<grid, THREADS, 0, stream>>>(
        src, dst, (const int8_t*)(ws + woff[l]), sclBase + soff[l],
        biaBase + soff[l], L_H[l], L_W[l], L_ICp[l], L_OC[l], L_OICp[l],
        L_POOL[l]);
  }

  fc_kernel<<<dim3(256, 10), 128, 0, stream>>>(bufA, fcw, fcb, outp);
}
